// ImprovedGraphAutoEncoder_82394652606590
// MI455X (gfx1250) — matrix operand order repaired and checked by reading, not compiled, not run on hardware
//
#include <hip/hip_runtime.h>
#include <math.h>

// ImprovedGraphAutoEncoder for MI455X (gfx1250, wave32, WMMA).
// B=8192 independent 8-node graphs. One wave handles 2 graphs (16 rows);
// 16x64 @ 64x64 GEMMs run on v_wmma_f32_16x16x32_f16 with LDS-resident,
// fragment-pre-swizzled f16 weights. Everything else is wave32 VALU with
// cross-lane shfl reductions. Single HBM pass (compute bound workload).

#define NEG_SLOPE 0.2f
#define ALPHA_RES 0.1f

typedef __attribute__((ext_vector_type(16))) _Float16 v16h;
typedef __attribute__((ext_vector_type(8)))  float    v8f;

// ---- WMMA fragment address helpers (halfword index inside a 16x64 A region,
// stored as 2 K-chunks of 1024B fragments, lane-major: lane*32B + vgpr*4B) ----
// 16-bit A matrix 16x32 layout (ISA 7.12.2):
//  lanes 0-15 : M=0..15, VGPR0..3 K=0..7, VGPR4..7 K=16..23
//  lanes 16-31: M=0..15, VGPR0..3 K=8..15, VGPR4..7 K=24..31
__device__ __forceinline__ int a_off_h(int m, int k) {
  int kc = k >> 5, kk = k & 31;
  int ln = (m & 15) + (((kk & 15) >= 8) ? 16 : 0);
  int v  = ((kk & 16) ? 4 : 0) + ((kk & 7) >> 1);
  return kc * 512 + ln * 16 + v * 2 + (kk & 1);
}

// 16-bit B matrix 32x16: lane = n + 16*((k&15)>=8); element s<8 holds K=8h+s, s>=8 holds K=16+8h+(s-8).
// Full 64x64 weight = [ntile(4)][kchunk(2)] fragments of 512 halves each.
__device__ __forceinline__ int b_frag_h(int k, int n) {
  int nt = n >> 4, kc = k >> 5;
  int kk = k & 31, nn = n & 15;
  int ln = nn + (((kk & 15) >= 8) ? 16 : 0);
  return (nt * 2 + kc) * 512 + ln * 16 + ((kk & 16) ? 8 : 0) + (kk & 7);
}

// Stage a 64x64 row-major f32 weight into LDS as f16 WMMA B-fragments.
__device__ __forceinline__ void stageB(const float* __restrict__ W,
                                       _Float16* dst, int tid, int nthr) {
  for (int i = tid; i < 4096; i += nthr) {
    int k = i >> 6, n = i & 63;
    dst[b_frag_h(k, n)] = (_Float16)W[i];
  }
}

// D = relu?(A(16x64) @ B(64x64) + bias). A in LDS A-fragment layout,
// Bf staged fragments, result stored back as A-fragments for the next layer.
__device__ __forceinline__ void mm64(const _Float16* A, const _Float16* Bf,
                                     const float* __restrict__ bias,
                                     bool relu, _Float16* Dst, int lane) {
  const v16h a0 = *(const v16h*)(A + lane * 16);
  const v16h a1 = *(const v16h*)(A + 512 + lane * 16);
#pragma unroll
  for (int nt = 0; nt < 4; ++nt) {
    const v16h b0 = *(const v16h*)(Bf + (nt * 2 + 0) * 512 + lane * 16);
    const v16h b1 = *(const v16h*)(Bf + (nt * 2 + 1) * 512 + lane * 16);
    v8f c = {};
    c = __builtin_amdgcn_wmma_f32_16x16x32_f16(false, a0, false, b0,
                                               (short)0, c, false, false);
    c = __builtin_amdgcn_wmma_f32_16x16x32_f16(false, a1, false, b1,
                                               (short)0, c, false, false);
    // C/D layout: lane l, VGPR v -> M = v + 8*(l>=16), N = nt*16 + (l&15)
    int mb = (lane >> 4) * 8;
    int n = nt * 16 + (lane & 15);
    float bn = bias[n];
#pragma unroll
    for (int v = 0; v < 8; ++v) {
      float val = c[v] + bn;
      if (relu) val = fmaxf(val, 0.0f);
      Dst[a_off_h(mb + v, n)] = (_Float16)val;
    }
  }
}

// Dense GATv2 (heads=1, edge_dim=1) over one wave's 2 graphs, d=64.
// Lane -> (e = lane>>4, i = (lane>>1)&7, 4 j's = (lane&1)*4..+3):
// each lane's 4 pairs share xr(i,:) so its LDS read is amortized 4x.
__device__ __forceinline__ void gat64(const _Float16* Xl, const _Float16* Xr,
                                      const float* adjB, const float* eaB,
                                      const float* __restrict__ att,
                                      const float* __restrict__ We,
                                      const float* __restrict__ bvec,
                                      float* alphaB, _Float16* Out,
                                      bool relu, int lane) {
  int e = lane >> 4, i = (lane >> 1) & 7, jb = (lane & 1) * 4;
  int ri = e * 8 + i;
  float s[4] = {0.f, 0.f, 0.f, 0.f};
  float eav[4];
#pragma unroll
  for (int q = 0; q < 4; ++q) eav[q] = eaB[e * 64 + i * 8 + jb + q];
  for (int d = 0; d < 64; ++d) {
    float xr = (float)Xr[a_off_h(ri, d)];
    float ad = att[d], we = We[d];   // d uniform -> scalar (SMEM) loads
#pragma unroll
    for (int q = 0; q < 4; ++q) {
      float z = xr + (float)Xl[a_off_h(e * 8 + jb + q, d)] + eav[q] * we;
      z = (z > 0.f) ? z : NEG_SLOPE * z;
      s[q] += ad * z;
    }
  }
  float mx = -3.0e38f;
#pragma unroll
  for (int q = 0; q < 4; ++q) {
    s[q] = (adjB[e * 64 + i * 8 + jb + q] > 0.5f) ? s[q] : -1.0e9f;
    mx = fmaxf(mx, s[q]);
  }
  mx = fmaxf(mx, __shfl_xor(mx, 1));       // partner lane holds other 4 j's
  float sum = 0.f;
#pragma unroll
  for (int q = 0; q < 4; ++q) { s[q] = __expf(s[q] - mx); sum += s[q]; }
  sum += __shfl_xor(sum, 1);
  float inv = 1.0f / sum;
#pragma unroll
  for (int q = 0; q < 4; ++q) alphaB[e * 64 + i * 8 + jb + q] = s[q] * inv;
  __syncthreads();
  // out(i,:) = sum_j alpha_ij * xl(j,:) + b ; 1024 outputs over 32 lanes
  for (int t = lane; t < 1024; t += 32) {
    int m = t >> 6, n = t & 63;
    int ee = m >> 3;
    float acc = bvec[n];
#pragma unroll
    for (int j = 0; j < 8; ++j)
      acc += alphaB[ee * 64 + (m & 7) * 8 + j] *
             (float)Xl[a_off_h(ee * 8 + j, n)];
    if (relu) acc = fmaxf(acc, 0.f);
    Out[a_off_h(m, n)] = (_Float16)acc;
  }
}

__global__ __launch_bounds__(128) void gae_kernel(
    const float* __restrict__ xin,
    const float* __restrict__ W1, const float* __restrict__ b1,
    const float* __restrict__ W2, const float* __restrict__ b2,
    const float* __restrict__ W3, const float* __restrict__ b3,
    const float* __restrict__ g1_Wl, const float* __restrict__ g1_bl,
    const float* __restrict__ g1_Wr, const float* __restrict__ g1_br,
    const float* __restrict__ g1_att, const float* __restrict__ g1_We,
    const float* __restrict__ g1_b,
    const float* __restrict__ g2_Wl, const float* __restrict__ g2_bl,
    const float* __restrict__ g2_Wr, const float* __restrict__ g2_br,
    const float* __restrict__ g2_att, const float* __restrict__ g2_We,
    const float* __restrict__ g2_b,
    const float* __restrict__ g3_Wl, const float* __restrict__ g3_bl,
    const float* __restrict__ g3_Wr, const float* __restrict__ g3_br,
    const float* __restrict__ g3_att, const float* __restrict__ g3_We,
    const float* __restrict__ g3_b,
    const float* __restrict__ Ws, const float* __restrict__ bs,
    float* __restrict__ out, int Btot) {
  // ---- shared: block-wide f16 weight fragments + per-wave scratch ----
  __shared__ alignas(32) _Float16 sW2[4096];
  __shared__ alignas(32) _Float16 sG2Wl[4096];
  __shared__ alignas(32) _Float16 sG2Wr[4096];
  __shared__ alignas(32) _Float16 sActA[4][1024];   // activation (A-frags)
  __shared__ alignas(32) _Float16 sActXl[4][1024];  // xl / ping-pong
  __shared__ alignas(32) _Float16 sActXr[4][1024];  // xr
  __shared__ float sLat[4][64];     // latent 16 x 3 (stride 4)
  __shared__ float sAdj[4][128];    // adj  2 x 8 x 8
  __shared__ float sEa[4][128];     // edge attr
  __shared__ float sAlpha[4][128];  // softmax weights
  __shared__ float sX[4][16];       // input values per row
  __shared__ float sXl3[4][48];     // g3 projections (16 x 3)
  __shared__ float sXr3[4][48];

  const int tid = threadIdx.x;
  const int w = tid >> 5, lane = tid & 31;
  const int e0 = (blockIdx.x * 4 + w) * 2;  // first of 2 graphs for this wave

  // ---- stage 64x64 weights into WMMA B-fragment layout (once per block) ----
  stageB(W2, sW2, tid, 128);
  stageB(g2_Wl, sG2Wl, tid, 128);
  stageB(g2_Wr, sG2Wr, tid, 128);

  // ---- load x, emit `pre` output ----
  if (lane < 16) {
    int e = e0 + (lane >> 3), node = lane & 7;
    float xv = xin[e * 8 + node];
    sX[w][lane] = xv;
    float* preo = out + (size_t)e * 24 + node * 3;
    preo[0] = 0.f; preo[1] = (float)node; preo[2] = xv;
  }
  __syncthreads();

  // ---- MLP layer 1: relu(pre @ W1 + b1), K=3 so scalar FMA is cheapest ----
  for (int t = lane; t < 1024; t += 32) {
    int m = t >> 6, n = t & 63;
    float acc = b1[n] + W1[64 + n] * (float)(m & 7) + W1[128 + n] * sX[w][m];
    sActA[w][a_off_h(m, n)] = (_Float16)fmaxf(acc, 0.f);
  }
  __syncthreads();

  // ---- MLP layer 2: 16x64 @ 64x64 on WMMA ----
  mm64(sActA[w], sW2, b2, true, sActXl[w], lane);
  __syncthreads();

  // ---- latent = h @ W3 + b3 (16x3), write `latent` output ----
  for (int t = lane; t < 48; t += 32) {
    int m = t / 3, c = t % 3;
    float acc = b3[c];
    for (int k = 0; k < 64; ++k)
      acc += (float)sActXl[w][a_off_h(m, k)] * W3[k * 3 + c];
    sLat[w][m * 4 + c] = acc;
    int e = e0 + (m >> 3), node = m & 7;
    out[(size_t)Btot * 48 + (size_t)e * 24 + node * 3 + c] = acc;
  }
  __syncthreads();

  // ---- Gabriel graph: 128 (e,i,j) pairs over 32 lanes, 4 per lane ----
  {
    int e = lane >> 4, i = (lane >> 1) & 7, jb = (lane & 1) * 4;
    const float* L = sLat[w];
    float pi0 = L[(e * 8 + i) * 4 + 0];
    float pi1 = L[(e * 8 + i) * 4 + 1];
    float pi2 = L[(e * 8 + i) * 4 + 2];
    float dist[4]; int gab[4];
    float lsum = 0.f; int lcnt = 0;
#pragma unroll
    for (int q = 0; q < 4; ++q) {
      int j = jb + q;
      float qx = L[(e * 8 + j) * 4 + 0];
      float qy = L[(e * 8 + j) * 4 + 1];
      float qz = L[(e * 8 + j) * 4 + 2];
      float dx = pi0 - qx, dy = pi1 - qy, dz = pi2 - qz;
      float d2 = dx * dx + dy * dy + dz * dz;
      int g = 0;
      if (j != i) {
        float mx = (pi0 + qx) * 0.5f, my = (pi1 + qy) * 0.5f,
              mz = (pi2 + qz) * 0.5f;
        int viol = 0;
        for (int k = 0; k < 8; ++k) {
          if (k == i || k == j) continue;
          float ax = L[(e * 8 + k) * 4 + 0] - mx;
          float ay = L[(e * 8 + k) * 4 + 1] - my;
          float az = L[(e * 8 + k) * 4 + 2] - mz;
          viol |= ((ax * ax + ay * ay + az * az) < 0.25f * d2) ? 1 : 0;
        }
        g = viol ? 0 : 1;
      }
      gab[q] = g;
      dist[q] = sqrtf(d2);
      if (g) { lsum += dist[q]; lcnt += 1; }
    }
    // reduce over this graph's 16 lanes (xor<16 stays within the half)
#pragma unroll
    for (int msk = 1; msk < 16; msk <<= 1) {
      lsum += __shfl_xor(lsum, msk);
      lcnt += __shfl_xor(lcnt, msk);
    }
    float mean = lsum / (float)(lcnt > 0 ? lcnt : 1);
#pragma unroll
    for (int q = 0; q < 4; ++q) {
      int j = jb + q;
      float eav = (gab[q] ? dist[q] : 0.f) + ((j == i) ? mean : 0.f);
      float adjv = (gab[q] || j == i) ? 1.f : 0.f;
      sEa[w][e * 64 + i * 8 + j] = eav;
      sAdj[w][e * 64 + i * 8 + j] = adjv;
      out[(size_t)Btot * 72 + (size_t)(e0 + e) * 64 + i * 8 + j] = adjv;
    }
  }
  __syncthreads();

  // ---- GAT layer 1 (din=1): xl/xr are rank-1, scalar ----
  for (int t = lane; t < 1024; t += 32) {
    int m = t >> 6, n = t & 63;
    float f = sLat[w][m * 4 + 2];  // feats = latent[:, :, 2]
    sActXl[w][a_off_h(m, n)] = (_Float16)(f * g1_Wl[n] + g1_bl[n]);
    sActXr[w][a_off_h(m, n)] = (_Float16)(f * g1_Wr[n] + g1_br[n]);
  }
  __syncthreads();
  gat64(sActXl[w], sActXr[w], sAdj[w], sEa[w], g1_att, g1_We, g1_b,
        sAlpha[w], sActA[w], true, lane);
  __syncthreads();

  // ---- GAT layer 2 (64->64): projections on WMMA ----
  mm64(sActA[w], sG2Wl, g2_bl, false, sActXl[w], lane);
  mm64(sActA[w], sG2Wr, g2_br, false, sActXr[w], lane);
  __syncthreads();
  gat64(sActXl[w], sActXr[w], sAdj[w], sEa[w], g2_att, g2_We, g2_b,
        sAlpha[w], sActA[w], true, lane);
  __syncthreads();

  // ---- GAT layer 3 (64->3): scalar projections ----
  for (int t = lane; t < 48; t += 32) {
    int m = t / 3, c = t % 3;
    float al = g3_bl[c], ar = g3_br[c];
    for (int k = 0; k < 64; ++k) {
      float xv = (float)sActA[w][a_off_h(m, k)];
      al += xv * g3_Wl[k * 3 + c];
      ar += xv * g3_Wr[k * 3 + c];
    }
    sXl3[w][m * 3 + c] = al;
    sXr3[w][m * 3 + c] = ar;
  }
  __syncthreads();
  {
    int e = lane >> 4, i = (lane >> 1) & 7, jb = (lane & 1) * 4;
    float s[4];
#pragma unroll
    for (int q = 0; q < 4; ++q) {
      int j = jb + q;
      float eav = sEa[w][e * 64 + i * 8 + j];
      float acc = 0.f;
#pragma unroll
      for (int d = 0; d < 3; ++d) {
        float z = sXr3[w][(e * 8 + i) * 3 + d] +
                  sXl3[w][(e * 8 + j) * 3 + d] + eav * g3_We[d];
        z = (z > 0.f) ? z : NEG_SLOPE * z;
        acc += g3_att[d] * z;
      }
      s[q] = (sAdj[w][e * 64 + i * 8 + j] > 0.5f) ? acc : -1.0e9f;
    }
    float mx = fmaxf(fmaxf(s[0], s[1]), fmaxf(s[2], s[3]));
    mx = fmaxf(mx, __shfl_xor(mx, 1));
    float sum = 0.f;
#pragma unroll
    for (int q = 0; q < 4; ++q) { s[q] = __expf(s[q] - mx); sum += s[q]; }
    sum += __shfl_xor(sum, 1);
    float inv = 1.0f / sum;
#pragma unroll
    for (int q = 0; q < 4; ++q)
      sAlpha[w][e * 64 + i * 8 + jb + q] = s[q] * inv;
  }
  __syncthreads();

  // ---- x3 aggregate + residual recon, write `recon` output ----
  for (int t = lane; t < 48; t += 32) {
    int m = t / 3, c = t % 3;
    int e = m >> 3;
    float acc = g3_b[c];
#pragma unroll
    for (int j = 0; j < 8; ++j)
      acc += sAlpha[w][e * 64 + (m & 7) * 8 + j] * sXl3[w][(e * 8 + j) * 3 + c];
    float lin = bs[c];
#pragma unroll
    for (int cc = 0; cc < 3; ++cc)
      lin += sLat[w][m * 4 + cc] * Ws[cc * 3 + c];
    int eg = e0 + e, node = m & 7;
    out[(size_t)Btot * 24 + (size_t)eg * 24 + node * 3 + c] =
        acc + ALPHA_RES * lin;
  }
}

extern "C" void kernel_launch(void* const* d_in, const int* in_sizes, int n_in,
                              void* d_out, int out_size, void* d_ws,
                              size_t ws_size, hipStream_t stream) {
  const float* p[30];
  for (int i = 0; i < 30; ++i) p[i] = (const float*)d_in[i];
  int B = in_sizes[0] / 8;  // 8192
  dim3 grid(B / 8);         // 4 waves/block, 2 graphs/wave -> 8 graphs/block
  dim3 block(128);
  gae_kernel<<<grid, block, 0, stream>>>(
      p[0], p[1], p[2], p[3], p[4], p[5], p[6], p[7], p[8], p[9], p[10], p[11],
      p[12], p[13], p[14], p[15], p[16], p[17], p[18], p[19], p[20], p[21],
      p[22], p[23], p[24], p[25], p[26], p[27], p[28], p[29], (float*)d_out, B);
}
